// SpacetimeNonLocalBlock_88605175316844
// MI455X (gfx1250) — matrix operand order repaired and checked by reading, not compiled, not run on hardware
//
#include <hip/hip_runtime.h>
#include <hip/hip_bf16.h>
#include <stdint.h>

// Problem constants (match reference)
#define BB 4
#define CC 256
#define ICC 128
#define NN 6272          // 8*28*28, divisible by 64
#define NT (NN / 16)     // 392 n-tiles
#define MCHUNK 64        // m columns processed per attention iteration

typedef __attribute__((ext_vector_type(16))) __bf16 v16bf;
typedef __attribute__((ext_vector_type(2)))  __bf16 v2bf;
typedef __attribute__((ext_vector_type(8)))  float  v8f;
typedef unsigned short ushortx;

union V16U {
    v16bf v;
    uint4 q[2];
};

__device__ __forceinline__ ushortx f2bf(float f) {
    unsigned u = __builtin_bit_cast(unsigned, f);
    unsigned r = u + 0x7FFFu + ((u >> 16) & 1u);   // round-to-nearest-even
    return (ushortx)(r >> 16);
}
__device__ __forceinline__ unsigned pack2(float lo, float hi) {
#if __has_builtin(__builtin_amdgcn_cvt_pk_bf16_f32)
    v2bf r = __builtin_amdgcn_cvt_pk_bf16_f32(lo, hi);
    return __builtin_bit_cast(unsigned, r);
#else
    return (unsigned)f2bf(lo) | ((unsigned)f2bf(hi) << 16);
#endif
}
__device__ __forceinline__ v16bf load32B(const ushortx* p) {
    V16U t;
    t.q[0] = ((const uint4*)p)[0];
    t.q[1] = ((const uint4*)p)[1];
    return t.v;
}
__device__ __forceinline__ v16bf load_pair(const ushortx* p0, const ushortx* p1) {
    V16U t;
    t.q[0] = *(const uint4*)p0;
    t.q[1] = *(const uint4*)p1;
    return t.v;
}
__device__ __forceinline__ v8f wmma_bf16(v16bf a, v16bf b, v8f c) {
    return __builtin_amdgcn_wmma_f32_16x16x32_bf16(false, a, false, b, (short)0, c, false, false);
}

// ---------------- prep: x[b][c][n] (f32) -> xb[b][n][c] (bf16) ----------------
__global__ void k_prep_x(const float* __restrict__ x, ushortx* __restrict__ xb) {
    size_t i = (size_t)blockIdx.x * blockDim.x + threadIdx.x;
    if (i >= (size_t)BB * CC * NN) return;
    int n = (int)(i % NN);
    size_t t = i / NN;
    int c = (int)(t % CC);
    int b = (int)(t / CC);
    xb[((size_t)b * NN + n) * CC + c] = f2bf(x[i]);
}

// ---------------- prep: 4 weight matrices f32 -> bf16 (each 128*256 elems) ----
__global__ void k_prep_w(const float* __restrict__ wg, const float* __restrict__ wth,
                         const float* __restrict__ wph, const float* __restrict__ wz,
                         ushortx* __restrict__ wgb, ushortx* __restrict__ wthb,
                         ushortx* __restrict__ wphb, ushortx* __restrict__ wzb) {
    int i = blockIdx.x * blockDim.x + threadIdx.x;
    if (i < ICC * CC) {
        wgb[i]  = f2bf(wg[i]);
        wthb[i] = f2bf(wth[i]);
        wphb[i] = f2bf(wph[i]);
        wzb[i]  = f2bf(wz[i]);
    }
}

// ---------------- projections: out[o][n] = sum_c W[o][c] * x[c][n] ------------
// grid (NT, B*3), 32 threads (1 wave). pj=0 -> g ([b][ic][n]); pj=1/2 -> theta/phi ([b][n][ic])
__global__ __launch_bounds__(32)
void k_proj(const ushortx* __restrict__ xb, const ushortx* __restrict__ w0,
            const ushortx* __restrict__ w1, const ushortx* __restrict__ w2,
            ushortx* __restrict__ g, ushortx* __restrict__ th, ushortx* __restrict__ ph) {
    const int lane = threadIdx.x;
    const int lo16 = lane & 15, hi = lane >> 4;
    const int n0 = blockIdx.x * 16;
    const int b  = blockIdx.y / 3;
    const int pj = blockIdx.y % 3;
    const ushortx* W = (pj == 0) ? w0 : ((pj == 1) ? w1 : w2);

    v8f acc[8] = {};
    const ushortx* xrow = xb + ((size_t)b * NN + n0 + lo16) * CC + hi * 16;
    for (int k4 = 0; k4 < 8; ++k4) {                 // K = C = 256 = 8*32
        v16bf Bt = load32B(xrow + k4 * 32);
        for (int ot = 0; ot < 8; ++ot) {             // O = IC = 128 = 8*16
            const ushortx* wr = W + (size_t)(16 * ot + lo16) * CC + k4 * 32 + hi * 16;
            v16bf At = load_pair(wr, wr + 8);
            acc[ot] = wmma_bf16(At, Bt, acc[ot]);
        }
    }
    const int n = n0 + lo16;
    if (pj == 0) {
        ushortx* gp = g + (size_t)b * ICC * NN;      // [ic][n] scatter
        for (int ot = 0; ot < 8; ++ot) {
            int icb = 16 * ot + hi * 8;
            for (int r = 0; r < 8; ++r)
                gp[(size_t)(icb + r) * NN + n] = f2bf(acc[ot][r]);
        }
    } else {
        ushortx* O = ((pj == 1) ? th : ph) + (size_t)b * NN * ICC;
        for (int ot = 0; ot < 8; ++ot) {             // [n][ic] contiguous
            uint4 q;
            q.x = pack2(acc[ot][0], acc[ot][1]);
            q.y = pack2(acc[ot][2], acc[ot][3]);
            q.z = pack2(acc[ot][4], acc[ot][5]);
            q.w = pack2(acc[ot][6], acc[ot][7]);
            *(uint4*)(O + (size_t)n * ICC + 16 * ot + hi * 8) = q;
        }
    }
}

// ---------------- fused flash attention ---------------------------------------
// y[n][ic] = softmax_m(theta^T phi) @ g^T.  Workgroup = 8 waves sharing one
// phi/g stream: async-copied into double-buffered LDS (XOR-swizzled 16B chunks),
// each wave owns a 16-column n-tile. S^T tiles keep softmax lane-local.
__global__ __launch_bounds__(256)
void k_attn(const ushortx* __restrict__ g, const ushortx* __restrict__ th,
            const ushortx* __restrict__ ph, ushortx* __restrict__ y) {
    // phi chunk: 64 rows x 128 ic (16KB); g chunk: 128 ic x 64 m (16KB); x2 buffers
    __shared__ ushortx phiS[2][MCHUNK * ICC];   // 32 KB
    __shared__ ushortx gS[2][ICC * MCHUNK];     // 32 KB

    const int tid  = threadIdx.x;
    const int lane = tid & 31;
    const int wv   = tid >> 5;                  // wave id 0..7
    const int lo16 = lane & 15, hi = lane >> 4;
    const int b  = blockIdx.y;
    const int n0 = (blockIdx.x * 8 + wv) * 16;

    const ushortx* thb = th + (size_t)b * NN * ICC;
    const ushortx* phb = ph + (size_t)b * NN * ICC;
    const ushortx* gb  = g  + (size_t)b * ICC * NN;

    const unsigned phiBase[2] = { (unsigned)(uintptr_t)&phiS[0][0],
                                  (unsigned)(uintptr_t)&phiS[1][0] };
    const unsigned gBase[2]   = { (unsigned)(uintptr_t)&gS[0][0],
                                  (unsigned)(uintptr_t)&gS[1][0] };

    // Async cooperative copy of one 64-m chunk into buffer bb (tracked by ASYNCcnt).
    auto issue = [&](int bbuf, int m0) {
        // phi rows m0..m0+63 are one contiguous 16KB block ([n][ic] layout).
        const char* psrc = (const char*)(phb + (size_t)m0 * ICC);
        for (int j = 0; j < 4; ++j) {
            unsigned c = (unsigned)tid + 256u * j;          // 16B chunk id 0..1023
            unsigned row = c >> 4, jj = c & 15;             // 16 chunks per 256B row
            unsigned lds = phiBase[bbuf] + (row * 16u + (jj ^ (row & 15u))) * 16u;
            uint64_t ga = (uint64_t)(psrc + c * 16u);
            asm volatile("global_load_async_to_lds_b128 %0, %1, off"
                         :: "v"(lds), "v"(ga) : "memory");
        }
        // g rows: ic 0..127, 128B each at gb + ic*NN + m0 (8 chunks per row).
        for (int j = 0; j < 4; ++j) {
            unsigned c = (unsigned)tid + 256u * j;
            unsigned ic = c >> 3, jj = c & 7;
            unsigned lds = gBase[bbuf] + (ic * 8u + (jj ^ (ic & 7u))) * 16u;
            uint64_t ga = (uint64_t)(const char*)(gb + (size_t)ic * NN + m0) + jj * 16u;
            asm volatile("global_load_async_to_lds_b128 %0, %1, off"
                         :: "v"(lds), "v"(ga) : "memory");
        }
    };

    // theta as B-operand (cols = n, K = ic), constant over the m loop
    v16bf Bth[4];
    {
        const ushortx* p = thb + (size_t)(n0 + lo16) * ICC + hi * 16;
        for (int k4 = 0; k4 < 4; ++k4) Bth[k4] = load32B(p + k4 * 32);
    }

    float rm = -3.0e38f, rl = 0.0f;
    v8f yacc[8] = {};
    const bool lolane = (lane < 16);

    issue(0, 0);
    int bbuf = 0;
    for (int m0 = 0; m0 < NN; m0 += MCHUNK) {
        asm volatile("s_wait_asynccnt 0x0" ::: "memory");
        __syncthreads();                               // buffer bbuf ready for all waves
        if (m0 + MCHUNK < NN) issue(bbuf ^ 1, m0 + MCHUNK);

        const ushortx* phB = (const ushortx*)(uintptr_t)0 + 0;  // (unused; keep simple)
        // ---- S^T subtiles s=0..3 : D[m][n] = sum_ic phi[m][ic] * theta^T[ic][n]
        v8f sv[4];
        for (int s = 0; s < 4; ++s) {
            v8f acc = {};
            const unsigned row = 16u * s + (unsigned)lo16;     // m-local row per lane
            const ushortx* rbase = &phiS[bbuf][row * ICC];
            for (int k4 = 0; k4 < 4; ++k4) {
                unsigned j0 = 4u * k4 + 2u * (unsigned)hi;     // logical 16B chunk
                unsigned j1 = j0 + 1u;
                const ushortx* p0 = rbase + (j0 ^ (row & 15u)) * 8u;   // 8 ushort = 16B
                const ushortx* p1 = rbase + (j1 ^ (row & 15u)) * 8u;
                v16bf A = load_pair(p0, p1);
                acc = wmma_bf16(A, Bth[k4], acc);
            }
            sv[s] = acc;
        }
        (void)phB;

        // ---- online softmax: lane owns column n0+lo16; partner lane^16 holds
        // the complementary 32 of this iteration's 64 m values.
        float vmax = sv[0][0];
        for (int s = 0; s < 4; ++s)
            for (int r = 0; r < 8; ++r) vmax = fmaxf(vmax, sv[s][r]);
        vmax = fmaxf(vmax, __shfl_xor(vmax, 16, 32));
        float newm = fmaxf(rm, vmax);
        float resc = __expf(rm - newm);
        rm = newm;
        float p[4][8];
        float ps = 0.0f;
        for (int s = 0; s < 4; ++s)
            for (int r = 0; r < 8; ++r) {
                p[s][r] = __expf(sv[s][r] - newm);
                ps += p[s][r];
            }
        ps += __shfl_xor(ps, 16, 32);
        rl = rl * resc + ps;
        for (int t = 0; t < 8; ++t)
            for (int r = 0; r < 8; ++r) yacc[t][r] *= resc;

        // ---- two K=32 P operands (chunk cch covers m-local 32cch..32cch+31)
        for (int cch = 0; cch < 2; ++cch) {
            const float* pp0 = p[2 * cch];
            const float* pp1 = p[2 * cch + 1];
            unsigned q0[4], q1[4], o0[4], o1[4];
            for (int j = 0; j < 4; ++j) {
                q0[j] = pack2(pp0[2 * j], pp0[2 * j + 1]);
                q1[j] = pack2(pp1[2 * j], pp1[2 * j + 1]);
            }
            for (int j = 0; j < 4; ++j) {
                o0[j] = __shfl_xor(q0[j], 16, 32);
                o1[j] = __shfl_xor(q1[j], 16, 32);
            }
            V16U bv;
            uint4 qa, qb;
            qa.x = lolane ? q0[0] : o1[0];  qa.y = lolane ? q0[1] : o1[1];
            qa.z = lolane ? q0[2] : o1[2];  qa.w = lolane ? q0[3] : o1[3];
            qb.x = lolane ? o0[0] : q1[0];  qb.y = lolane ? o0[1] : q1[1];
            qb.z = lolane ? o0[2] : q1[2];  qb.w = lolane ? o0[3] : q1[3];
            bv.q[0] = qa; bv.q[1] = qb;
            v16bf B2 = bv.v;

            // yacc[t] (rows = ic, cols = n) += g[ic][m-chunk] * P
            for (int t = 0; t < 8; ++t) {
                unsigned ic = 16u * t + (unsigned)lo16;
                const ushortx* rbase = &gS[bbuf][ic * MCHUNK];
                unsigned j0 = 4u * cch + 2u * (unsigned)hi;    // logical 16B chunk (of 8)
                unsigned j1 = j0 + 1u;                          // +8 bf16 = next chunk
                const ushortx* p0 = rbase + (j0 ^ (ic & 7u)) * 8u;
                const ushortx* p1 = rbase + (j1 ^ (ic & 7u)) * 8u;
                v16bf Ag = load_pair(p0, p1);
                yacc[t] = wmma_bf16(Ag, B2, yacc[t]);
            }
        }
        __syncthreads();                               // all waves done reading bbuf
        bbuf ^= 1;
    }

    const float inv = 1.0f / rl;
    ushortx* yb = y + (size_t)b * NN * ICC;
    const int n = n0 + lo16;
    for (int t = 0; t < 8; ++t) {
        uint4 q;
        q.x = pack2(yacc[t][0] * inv, yacc[t][1] * inv);
        q.y = pack2(yacc[t][2] * inv, yacc[t][3] * inv);
        q.z = pack2(yacc[t][4] * inv, yacc[t][5] * inv);
        q.w = pack2(yacc[t][6] * inv, yacc[t][7] * inv);
        *(uint4*)(yb + (size_t)n * ICC + 16 * t + hi * 8) = q;
    }
}

// ---------------- final: z[c][n] = sum_ic Wz[c][ic] * y[ic][n] + x[c][n] ------
__global__ __launch_bounds__(32)
void k_final(const float* __restrict__ x, const ushortx* __restrict__ wzb,
             const ushortx* __restrict__ y, float* __restrict__ out) {
    const int lane = threadIdx.x;
    const int lo16 = lane & 15, hi = lane >> 4;
    const int n0 = blockIdx.x * 16;
    const int b  = blockIdx.y;
    const int o0 = blockIdx.z * 128;

    v16bf Bt[4];
    const ushortx* yb = y + (size_t)b * NN * ICC + (size_t)(n0 + lo16) * ICC + hi * 16;
    for (int k4 = 0; k4 < 4; ++k4) Bt[k4] = load32B(yb + k4 * 32);

    v8f acc[8] = {};
    for (int ot = 0; ot < 8; ++ot) {
        const ushortx* wr = wzb + (size_t)(o0 + 16 * ot + lo16) * ICC + hi * 16;
        for (int k4 = 0; k4 < 4; ++k4) {
            v16bf At = load_pair(wr + k4 * 32, wr + k4 * 32 + 8);
            acc[ot] = wmma_bf16(At, Bt[k4], acc[ot]);
        }
    }
    const int n = n0 + lo16;
    for (int ot = 0; ot < 8; ++ot) {
        int cb = o0 + 16 * ot + hi * 8;
        for (int r = 0; r < 8; ++r) {
            size_t idx = ((size_t)b * CC + cb + r) * NN + n;
            out[idx] = acc[ot][r] + x[idx];
        }
    }
}

extern "C" void kernel_launch(void* const* d_in, const int* in_sizes, int n_in,
                              void* d_out, int out_size, void* d_ws, size_t ws_size,
                              hipStream_t stream) {
    const float* x   = (const float*)d_in[0];
    const float* wg  = (const float*)d_in[1];
    const float* wth = (const float*)d_in[2];
    const float* wph = (const float*)d_in[3];
    const float* wz  = (const float*)d_in[4];
    float* out = (float*)d_out;

    char* ws = (char*)d_ws;
    size_t off = 0;
    ushortx* xb   = (ushortx*)(ws + off); off += (size_t)BB * NN * CC * 2;
    ushortx* wgb  = (ushortx*)(ws + off); off += (size_t)ICC * CC * 2;
    ushortx* wthb = (ushortx*)(ws + off); off += (size_t)ICC * CC * 2;
    ushortx* wphb = (ushortx*)(ws + off); off += (size_t)ICC * CC * 2;
    ushortx* wzb  = (ushortx*)(ws + off); off += (size_t)CC * ICC * 2;
    ushortx* gq   = (ushortx*)(ws + off); off += (size_t)BB * ICC * NN * 2;
    ushortx* thq  = (ushortx*)(ws + off); off += (size_t)BB * NN * ICC * 2;
    ushortx* phq  = (ushortx*)(ws + off); off += (size_t)BB * NN * ICC * 2;
    ushortx* yq   = (ushortx*)(ws + off); off += (size_t)BB * NN * ICC * 2;

    {
        size_t total = (size_t)BB * CC * NN;
        int blocks = (int)((total + 255) / 256);
        k_prep_x<<<blocks, 256, 0, stream>>>(x, xb);
    }
    {
        int total = ICC * CC;
        k_prep_w<<<(total + 255) / 256, 256, 0, stream>>>(wg, wth, wph, wz,
                                                          wgb, wthb, wphb, wzb);
    }
    k_proj<<<dim3(NT, BB * 3), 32, 0, stream>>>(xb, wgb, wthb, wphb, gq, thq, phq);
    k_attn<<<dim3(NT / 8, BB), 256, 0, stream>>>(gq, thq, phq, yq);
    k_final<<<dim3(NT, BB, 2), 32, 0, stream>>>(x, wzb, yq, out);
}
